// QuantLinear_81174881894900
// MI455X (gfx1250) — matrix operand order repaired and checked by reading, not compiled, not run on hardware
//
#include <hip/hip_runtime.h>

// ---------------------------------------------------------------------------
// GPTQ 4-bit (group=128) quantized linear:  out = x @ dequant(qweight) + bias
//   x:       [8192, 4096]  fp32
//   qweight: [4096/8, 11008] int32 (8 K-nibbles per word, column-packed)
//   qzeros:  [4096/128, 11008/8] int32
//   scales:  [4096/128, 11008] fp32
//   out:     [8192, 11008] fp32
// Strategy: dequant-to-LDS (f16) + V_WMMA_F32_16X16X32_F16 with f32 accum.
// BK=64: 16 WMMAs per barrier round, tile never straddles a quant group.
// ---------------------------------------------------------------------------

typedef __attribute__((ext_vector_type(16))) _Float16 v16h;
typedef __attribute__((ext_vector_type(8)))  float    v8f;

#define IN_F   4096
#define OUT_F  11008
#define GROUPK 128

#define BM 128
#define BN 128
#define BK 64
#define BKP 72   // padded LDS row (144 B): 16B-aligned b128 slots, skewed banks

union H8  { float4 f;    _Float16 h[8];  };
union H16 { float4 f[2]; _Float16 h[16]; };
union FragU { v16h v; float4 q[2]; };

__global__ __launch_bounds__(256)
void gptq_gemm_f16_wmma(const float* __restrict__ x,
                        const int*   __restrict__ qweight,
                        const int*   __restrict__ qzeros,
                        const float* __restrict__ scales,
                        const float* __restrict__ bias,
                        float* __restrict__ out)
{
    __shared__ _Float16 sA[BM][BKP];   // x tile, f16, row-major (M x K)
    __shared__ _Float16 sB[BN][BKP];   // W tile, f16, transposed (N x K)

    const int tid  = threadIdx.x;
    const int lane = tid & 31;
    const int wave = tid >> 5;
    const int wm   = wave >> 1;        // 0..3 : 32-row slab of the block tile
    const int wn   = wave & 1;         // 0..1 : 64-col slab

    const int n0 = blockIdx.x * BN;
    const int m0 = blockIdx.y * BM;

    // --- staging map: X (each thread: 32 consecutive floats of one row) ---
    const int xr = tid >> 1;
    const int xc = (tid & 1) * 32;
    const float* xrow = x + (size_t)(m0 + xr) * IN_F + xc;

    // --- staging map: W (each thread: one column, four 8-K chunks) ---
    const int  wcol   = tid & 127;            // column within tile
    const int  kcBase = (tid >> 7) * 4;       // first 8-K chunk index (0 or 4)
    const int  gcol   = n0 + wcol;            // global column
    const int  zshift = (gcol & 7) * 4;
    const int* qwcol  = qweight + gcol;       // stride OUT_F per packed K-row

    v8f acc[2][4] = {};                       // 8 x (16x16 f32) accumulators

    const int halfSel = lane >> 4;            // 0: lanes 0-15, 1: lanes 16-31
    const int lidx    = lane & 15;

    for (int k0 = 0; k0 < IN_F; k0 += BK) {
        // ---- prefetch next K-tile into near caches while computing ----
        if (k0 + BK < IN_F) {
            __builtin_prefetch(xrow + k0 + BK, 0, 3);
            __builtin_prefetch(qwcol + (size_t)((k0 + BK) >> 3) * OUT_F, 0, 3);
        }

        // ---- stage X tile: fp32 global -> f16 LDS ----
        {
            const float4* p = (const float4*)(xrow + k0);
            #pragma unroll
            for (int h = 0; h < 2; ++h) {
                H16 hv;
                #pragma unroll
                for (int j = 0; j < 4; ++j) {
                    float4 v = p[h * 4 + j];
                    hv.h[4*j+0] = (_Float16)v.x;
                    hv.h[4*j+1] = (_Float16)v.y;
                    hv.h[4*j+2] = (_Float16)v.z;
                    hv.h[4*j+3] = (_Float16)v.w;
                }
                *(float4*)&sA[xr][xc + h*16]     = hv.f[0];
                *(float4*)&sA[xr][xc + h*16 + 8] = hv.f[1];
            }
        }

        // ---- dequantize W tile: int4 -> f16 LDS (transposed N x K) ----
        {
            const int   g = k0 >> 7;  // GROUPK=128; BK=64 never straddles
            const float s = scales[(size_t)g * OUT_F + gcol];
            const int   z = (qzeros[(size_t)g * (OUT_F / 8) + (gcol >> 3)]
                             >> zshift) & 15;
            const float nzs = -s * (float)z;   // w = fma(q, s, -s*z)
            #pragma unroll
            for (int j = 0; j < 4; ++j) {
                const int kk = (kcBase + j) * 8;
                const int w  = qwcol[(size_t)((k0 + kk) >> 3) * OUT_F];
                H8 hb;
                #pragma unroll
                for (int b = 0; b < 8; ++b) {
                    int q = (w >> (4 * b)) & 15;
                    hb.h[b] = (_Float16)fmaf((float)q, s, nzs);
                }
                *(float4*)&sB[wcol][kk] = hb.f;
            }
        }

        __syncthreads();

        // ---- two WMMA K-substeps per staged tile: 16 matrix ops/round ----
        #pragma unroll
        for (int ks = 0; ks < 2; ++ks) {
            const int kb = ks * 32;

            // A 16x32: lanes 0-15 -> K{0-7,16-23}, lanes 16-31 -> K{8-15,24-31}
            FragU a[2];
            #pragma unroll
            for (int i = 0; i < 2; ++i) {
                const _Float16* ap =
                    &sA[wm * 32 + i * 16 + lidx][kb + halfSel * 8];
                a[i].q[0] = *(const float4*)(ap);        // K base .. +7
                a[i].q[1] = *(const float4*)(ap + 16);   // K base+16 .. +23
            }
            // B 32x16: lanes 0-15 -> K{0-7,16-23}, lanes 16-31 -> K{8-15,24-31}
            FragU bfr[4];
            #pragma unroll
            for (int j = 0; j < 4; ++j) {
                const _Float16* bp =
                    &sB[wn * 64 + j * 16 + lidx][kb + halfSel * 8];
                bfr[j].q[0] = *(const float4*)(bp);
                bfr[j].q[1] = *(const float4*)(bp + 16);
            }

            #pragma unroll
            for (int i = 0; i < 2; ++i)
                #pragma unroll
                for (int j = 0; j < 4; ++j)
                    acc[i][j] = __builtin_amdgcn_wmma_f32_16x16x32_f16(
                        false, a[i].v, false, bfr[j].v,
                        (short)0, acc[i][j], false, false);
        }

        __syncthreads();
    }

    // ---- epilogue: C layout VGPR e -> M = half*8+e, N = lane&15 ----
    #pragma unroll
    for (int j = 0; j < 4; ++j) {
        const int   col = n0 + wn * 64 + j * 16 + lidx;
        const float bv  = bias[col];
        #pragma unroll
        for (int i = 0; i < 2; ++i) {
            const int rbase = m0 + wm * 32 + i * 16 + halfSel * 8;
            #pragma unroll
            for (int e = 0; e < 8; ++e) {
                out[(size_t)(rbase + e) * OUT_F + col] = acc[i][j][e] + bv;
            }
        }
    }
}

extern "C" void kernel_launch(void* const* d_in, const int* in_sizes, int n_in,
                              void* d_out, int out_size, void* d_ws, size_t ws_size,
                              hipStream_t stream) {
    const float* x       = (const float*)d_in[0];
    const int*   qweight = (const int*)  d_in[1];
    const int*   qzeros  = (const int*)  d_in[2];
    const float* scales  = (const float*)d_in[3];
    const float* bias    = (const float*)d_in[4];
    float*       out     = (float*)d_out;

    const int M = in_sizes[0] / IN_F;          // 8192
    dim3 grid(OUT_F / BN, M / BM);             // 86 x 64
    gptq_gemm_f16_wmma<<<grid, 256, 0, stream>>>(x, qweight, qzeros, scales,
                                                 bias, out);
    (void)n_in; (void)out_size; (void)d_ws; (void)ws_size;
}
